// PRXAttention_51771535786297
// MI455X (gfx1250) — matrix operand order repaired and checked by reading, not compiled, not run on hardware
//
#include <hip/hip_runtime.h>
#include <hip/hip_bf16.h>

// ---------------------------------------------------------------------------
// MI455X (gfx1250) fused PRX joint attention.
// Compute-bound (~240 GFLOP vs ~175MB traffic) -> everything on
// v_wmma_f32_16x16x32_bf16 with fp32 accumulation, epilogues fused.
// Direct-copy LDS staging uses CDNA5 async global->LDS loads (ASYNCcnt).
// ---------------------------------------------------------------------------

typedef __attribute__((ext_vector_type(16))) __bf16 v16bf;
typedef __attribute__((ext_vector_type(8)))  float  v8f;
typedef __attribute__((ext_vector_type(4)))  int    v4i;

// Problem constants
#define BB   2
#define LI   2048
#define LT   512
#define DD   2048
#define HH   16
#define DH   128
#define LK   2560   // LT + LI

// ---- CDNA5 async global->LDS copy (guarded; falls back to sync copy) ------
#if __has_builtin(__builtin_amdgcn_global_load_async_to_lds_b128) && \
    __has_builtin(__builtin_amdgcn_s_wait_asynccnt)
#define USE_ASYNC_LDS 1
#else
#define USE_ASYNC_LDS 0
#endif

typedef __attribute__((address_space(1))) v4i gl_v4i;
typedef __attribute__((address_space(3))) v4i ld_v4i;

__device__ __forceinline__ void stage16(void* lds_dst, const void* gsrc) {
#if USE_ASYNC_LDS
  __builtin_amdgcn_global_load_async_to_lds_b128((gl_v4i*)gsrc,
                                                 (ld_v4i*)lds_dst, 0, 0);
#else
  *(uint4*)lds_dst = *(const uint4*)gsrc;
#endif
}

__device__ __forceinline__ void stage_wait() {
#if USE_ASYNC_LDS
  __builtin_amdgcn_s_wait_asynccnt(0);
#endif
}

__device__ __forceinline__ v8f wmma_bf16(v16bf a, v16bf b, v8f c) {
  // (neg_a, A, neg_b, B, c_mod, C, reuse_a, reuse_b)
  return __builtin_amdgcn_wmma_f32_16x16x32_bf16(false, a, false, b, (short)0, c,
                                                 false, false);
}

// A-matrix fragment (16x32 bf16) from row-major [M][ldk] storage.
// ISA 7.12.2: lanes 0-15 -> M=lane, K = {k0..k0+7, k0+16..k0+23};
//             lanes 16-31 -> M=lane-16, K shifted by +8.
__device__ __forceinline__ v16bf load_a_frag(const __bf16* base, int ldk,
                                             int mBase, int k0, int lane) {
  const int row = mBase + (lane & 15);
  const int kk  = k0 + ((lane & 16) >> 1);   // +8 for upper half-wave
  const __bf16* p = base + (size_t)row * ldk + kk;
  v16bf a;
#pragma unroll
  for (int j = 0; j < 8; ++j) { a[j] = p[j]; a[j + 8] = p[j + 16]; }
  return a;
}

// B-matrix fragment (32x16 bf16) from N-major [N][ldk] storage (K contiguous).
// lanes 0-15 -> N=lane, K = {k0..k0+7, k0+16..k0+23}; lanes 16-31 -> N=lane-16, K shifted by +8.
__device__ __forceinline__ v16bf load_b_frag(const __bf16* base, int ldk,
                                             int nBase, int k0, int lane) {
  const int row = nBase + (lane & 15);
  const int kk  = k0 + ((lane & 16) >> 1);   // +8 for upper half-wave
  const __bf16* p = base + (size_t)row * ldk + kk;
  v16bf b;
#pragma unroll
  for (int j = 0; j < 8; ++j) { b[j] = p[j]; b[j + 8] = p[j + 16]; }
  return b;
}

// ---------------------------------------------------------------------------
// f32 -> bf16 cast, vectorized by 4
// ---------------------------------------------------------------------------
__device__ __forceinline__ unsigned int f2bf_bits(float f) {
  __bf16 h = (__bf16)f;
  return (unsigned int)__builtin_bit_cast(unsigned short, h);
}

__global__ __launch_bounds__(256) void cast_f32_bf16(const float* __restrict__ in,
                                                     __bf16* __restrict__ out,
                                                     int n4) {
  const int i = blockIdx.x * 256 + threadIdx.x;
  if (i < n4) {
    const float4 v = ((const float4*)in)[i];
    uint2 o;
    o.x = f2bf_bits(v.x) | (f2bf_bits(v.y) << 16);
    o.y = f2bf_bits(v.z) | (f2bf_bits(v.w) << 16);
    ((uint2*)out)[i] = o;
  }
}

// ---------------------------------------------------------------------------
// Generic bf16 GEMM, block tile 64(M) x 128(N), K-step 32, 8 waves (2x4),
// wave tile 32x32 -> 4 v8f accumulators. lda is always D (=2048).
// MODE 0: img QKV epilogue (RMSNorm q/k + RoPE, scatter to Q/K/V bf16)
// MODE 1: txt KV epilogue  (RMSNorm k, scatter to K/V bf16 rows [0,512))
// MODE 2: plain fp32 store (final out-projection)
// ---------------------------------------------------------------------------
#define GBM 64
#define GBN 128
#define GBK 32
#define GLDK 40     // padded LDS K-stride (bf16): 80B rows, 16B-aligned
#define GLDC 132    // padded LDS C-stride (f32)

template <int MODE>
__global__ __launch_bounds__(256) void gemm_kernel(
    const __bf16* __restrict__ A, const __bf16* __restrict__ Bm, int ldb,
    __bf16* __restrict__ Qb, __bf16* __restrict__ Kb, __bf16* __restrict__ Vb,
    const float* __restrict__ fc, const float* __restrict__ g0,
    const float* __restrict__ g1, float* __restrict__ outf) {
  __shared__ __align__(16) unsigned char smem[GBM * GLDC * 4 + GBM * 4 * 4];
  __bf16* ldsA = (__bf16*)smem;                 // [GBM][GLDK]
  __bf16* ldsB = (__bf16*)smem + GBM * GLDK;    // [GBN][GLDK]  (N-major)

  const int tid  = threadIdx.x;
  const int lane = tid & 31;
  const int wave = tid >> 5;
  const int wm   = (wave >> 2) * 32;   // wave M offset in tile
  const int wn   = (wave & 3) * 32;    // wave N offset in tile
  const int mBase = blockIdx.x * GBM;
  const int nBase = blockIdx.y * GBN;

  v8f acc[2][2];
#pragma unroll
  for (int i = 0; i < 2; ++i)
#pragma unroll
    for (int j = 0; j < 2; ++j) acc[i][j] = v8f{};

  for (int kt = 0; kt < DD; kt += GBK) {
    // A tile: 64x32, 16B per thread, coalesced -> async global->LDS
    {
      const int r  = tid >> 2;
      const int cg = (tid & 3) * 8;
      stage16(ldsA + r * GLDK + cg, A + (size_t)(mBase + r) * DD + kt + cg);
    }
    // B tile: 32(K)x128(N) transposed into ldsB[N][K] (sync: reshuffles data)
#pragma unroll
    for (int g = 0; g < 2; ++g) {
      const int idx = tid + g * 256;
      const int kk  = idx >> 4;
      const int ng  = (idx & 15) * 8;
      const uint4 v =
          *(const uint4*)(Bm + (size_t)(kt + kk) * ldb + nBase + ng);
      const __bf16* e = (const __bf16*)&v;
#pragma unroll
      for (int j = 0; j < 8; ++j) ldsB[(ng + j) * GLDK + kk] = e[j];
    }
    stage_wait();
    __syncthreads();

    const v16bf a0 = load_a_frag(ldsA, GLDK, wm + 0, 0, lane);
    const v16bf a1 = load_a_frag(ldsA, GLDK, wm + 16, 0, lane);
    const v16bf b0 = load_b_frag(ldsB, GLDK, wn + 0, 0, lane);
    const v16bf b1 = load_b_frag(ldsB, GLDK, wn + 16, 0, lane);
    acc[0][0] = wmma_bf16(a0, b0, acc[0][0]);
    acc[0][1] = wmma_bf16(a0, b1, acc[0][1]);
    acc[1][0] = wmma_bf16(a1, b0, acc[1][0]);
    acc[1][1] = wmma_bf16(a1, b1, acc[1][1]);
    __syncthreads();
  }

  const int rh = (lane & 16) >> 1;   // +8 row offset for upper half-wave

  if constexpr (MODE == 2) {
    // plain fp32 store, C-fragment layout: row = r + rh, col = lane&15
#pragma unroll
    for (int i = 0; i < 2; ++i)
#pragma unroll
      for (int j = 0; j < 2; ++j)
#pragma unroll
        for (int r = 0; r < 8; ++r) {
          const int rowg = mBase + wm + 16 * i + r + rh;
          const int colg = nBase + wn + 16 * j + (lane & 15);
          outf[(size_t)rowg * DD + colg] = acc[i][j][r];
        }
    return;
  }

  // Stage C tile to LDS so each row's full 128-wide head vector is visible
  // for the fused RMSNorm / RoPE epilogue.
  __syncthreads();
  float* ldsC = (float*)smem;               // [GBM][GLDC]
  float* red  = (float*)smem + GBM * GLDC;  // [GBM][4] partial sum-squares
#pragma unroll
  for (int i = 0; i < 2; ++i)
#pragma unroll
    for (int j = 0; j < 2; ++j)
#pragma unroll
      for (int r = 0; r < 8; ++r)
        ldsC[(wm + 16 * i + r + rh) * GLDC + wn + 16 * j + (lane & 15)] =
            acc[i][j][r];
  __syncthreads();

  const int r    = tid >> 2;          // row in tile 0..63
  const int cseg = (tid & 3) * 32;    // 32-wide column segment
  const int row  = mBase + r;
  const float* crow = ldsC + r * GLDC;

  if constexpr (MODE == 0) {
    const int qkv = nBase >> 11;        // n = ((qkv*H)+h)*DH + d
    const int h   = (nBase >> 7) & 15;
    const int b   = row >> 11;
    const int l   = row & (LI - 1);
    if (qkv == 2) {  // V: plain bf16 cast, img rows live at kv index 512+l
      __bf16* dst = Vb + ((size_t)(b * HH + h) * LK + LT + l) * DH + cseg;
#pragma unroll
      for (int c = 0; c < 32; ++c) dst[c] = (__bf16)crow[cseg + c];
    } else {  // Q or K: RMSNorm + RoPE
      float ss = 0.f;
#pragma unroll
      for (int c = 0; c < 32; ++c) {
        const float x = crow[cseg + c];
        ss += x * x;
      }
      red[r * 4 + (tid & 3)] = ss;
      __syncthreads();
      const float var =
          (red[r * 4] + red[r * 4 + 1] + red[r * 4 + 2] + red[r * 4 + 3]) *
          (1.f / 128.f);
      const float rs = rsqrtf(var + 1e-6f);
      const float* g = (qkv == 0) ? g0 : g1;
      const float* fcb = fc + ((size_t)b * LI + l) * 256;  // [64][2][2] f32
      __bf16* dst = (qkv == 0)
                        ? Qb + ((size_t)(b * HH + h) * LI + l) * DH + cseg
                        : Kb + ((size_t)(b * HH + h) * LK + LT + l) * DH + cseg;
#pragma unroll
      for (int c = 0; c < 32; c += 2) {
        const int d0 = cseg + c;
        const float x0 = crow[d0] * rs * g[d0];
        const float x1 = crow[d0 + 1] * rs * g[d0 + 1];
        const float4 f = *(const float4*)(fcb + (d0 >> 1) * 4);
        dst[c]     = (__bf16)(f.x * x0 + f.y * x1);   // cos*x0 - sin*x1
        dst[c + 1] = (__bf16)(f.z * x0 + f.w * x1);   // sin*x0 + cos*x1
      }
    }
  } else {  // MODE == 1, txt KV
    const int kv = nBase >> 11;
    const int h  = (nBase >> 7) & 15;
    const int b  = row >> 9;
    const int l  = row & (LT - 1);
    __bf16* dst =
        ((kv == 0) ? Kb : Vb) + ((size_t)(b * HH + h) * LK + l) * DH + cseg;
    if (kv == 1) {
#pragma unroll
      for (int c = 0; c < 32; ++c) dst[c] = (__bf16)crow[cseg + c];
    } else {  // K: RMSNorm with g_added_k, no RoPE
      float ss = 0.f;
#pragma unroll
      for (int c = 0; c < 32; ++c) {
        const float x = crow[cseg + c];
        ss += x * x;
      }
      red[r * 4 + (tid & 3)] = ss;
      __syncthreads();
      const float var =
          (red[r * 4] + red[r * 4 + 1] + red[r * 4 + 2] + red[r * 4 + 3]) *
          (1.f / 128.f);
      const float rs = rsqrtf(var + 1e-6f);
#pragma unroll
      for (int c = 0; c < 32; ++c)
        dst[c] = (__bf16)(crow[cseg + c] * rs * g0[cseg + c]);
    }
  }
}

// ---------------------------------------------------------------------------
// Flash attention: block = 8 waves, each wave owns 16 query rows (128/block).
// KV walked in 64-row chunks; K staged [kv][dh] via async global->LDS
// (== B-frag [N][K] layout), V staged transposed [dh][kv]. Softmax stats
// live on the C-fragment half-wave layout, reduced with 16-lane shuffles.
// ---------------------------------------------------------------------------
#define AK_LD 136   // ldsK K-stride (bf16), 272B rows
#define AV_LD 80    // ldsVt K-stride (bf16), 160B rows

__global__ __launch_bounds__(256) void attn_kernel(
    const __bf16* __restrict__ Q, const __bf16* __restrict__ K,
    const __bf16* __restrict__ V, const unsigned char* __restrict__ mask,
    __bf16* __restrict__ attnb) {
  __shared__ __align__(16)
      unsigned char smem[(64 * AK_LD + DH * AV_LD + 8 * 16 * 64) * 2];
  __bf16* ldsK  = (__bf16*)smem;          // [64][AK_LD]
  __bf16* ldsVt = ldsK + 64 * AK_LD;      // [128][AV_LD]
  __bf16* ldsP  = ldsVt + DH * AV_LD;     // [8 waves][16][64]

  const int tid  = threadIdx.x;
  const int lane = tid & 31;
  const int wave = tid >> 5;
  const int ln   = lane & 15;
  const int rh   = (lane & 16) >> 1;
  const int bh   = blockIdx.y;
  const int b    = bh >> 4;
  const int h    = bh & 15;
  const int qBase = blockIdx.x * 128;

  const __bf16* Qp = Q + (size_t)bh * LI * DH;
  const __bf16* Kp = K + (size_t)bh * LK * DH;
  const __bf16* Vp = V + (size_t)bh * LK * DH;

  // Q fragments for this wave's 16 rows, resident for the whole KV sweep
  v16bf qf[4];
#pragma unroll
  for (int kc = 0; kc < 4; ++kc)
    qf[kc] =
        load_a_frag(Qp + (size_t)(qBase + wave * 16) * DH, DH, 0, kc * 32, lane);

  float mrow[8], lrow[8];
  v8f acc[8];
#pragma unroll
  for (int r = 0; r < 8; ++r) {
    mrow[r] = -__builtin_inff();
    lrow[r] = 0.f;
  }
#pragma unroll
  for (int j = 0; j < 8; ++j) acc[j] = v8f{};

  constexpr float SCALE = 0.08838834764831845f;  // 1/sqrt(128)
  constexpr float LOG2E = 1.4426950408889634f;

  for (int ck = 0; ck < LK / 64; ++ck) {
    const int kv0 = ck * 64;
    // stage K chunk: direct copy -> async global->LDS (16B per thread x4)
#pragma unroll
    for (int g = 0; g < 4; ++g) {
      const int idx = tid + g * 256;
      const int rw  = idx >> 4;
      const int cg  = (idx & 15) * 8;
      stage16(ldsK + rw * AK_LD + cg, Kp + (size_t)(kv0 + rw) * DH + cg);
    }
    // stage V chunk transposed -> [dh][kv] (sync: reshuffles data)
#pragma unroll
    for (int g = 0; g < 4; ++g) {
      const int idx = tid + g * 256;
      const int rw  = idx >> 4;
      const int cg  = (idx & 15) * 8;
      const uint4 v = *(const uint4*)(Vp + (size_t)(kv0 + rw) * DH + cg);
      const __bf16* e = (const __bf16*)&v;
#pragma unroll
      for (int j = 0; j < 8; ++j) ldsVt[(cg + j) * AV_LD + rw] = e[j];
    }
    stage_wait();
    __syncthreads();

    // scores: 16 x 64 per wave
    v8f sc[4];
#pragma unroll
    for (int n = 0; n < 4; ++n) {
      v8f c = v8f{};
#pragma unroll
      for (int kc = 0; kc < 4; ++kc)
        c = wmma_bf16(qf[kc], load_b_frag(ldsK, AK_LD, n * 16, kc * 32, lane), c);
      sc[n] = c;
    }

    // mask + scale + row max
    float rmax[8];
#pragma unroll
    for (int r = 0; r < 8; ++r) rmax[r] = -__builtin_inff();
#pragma unroll
    for (int n = 0; n < 4; ++n) {
      const int col = kv0 + n * 16 + ln;
      const bool valid = (col >= LT) || (mask[(size_t)b * LT + col] != 0);
#pragma unroll
      for (int r = 0; r < 8; ++r) {
        const float s = valid ? sc[n][r] * SCALE : -__builtin_inff();
        sc[n][r] = s;
        rmax[r] = fmaxf(rmax[r], s);
      }
    }
#pragma unroll
    for (int r = 0; r < 8; ++r)
#pragma unroll
      for (int off = 1; off < 16; off <<= 1)
        rmax[r] = fmaxf(rmax[r], __shfl_xor(rmax[r], off, 16));

    float corr[8], rsum[8];
#pragma unroll
    for (int r = 0; r < 8; ++r) {
      const float mn = fmaxf(mrow[r], rmax[r]);
      corr[r] = exp2f((mrow[r] - mn) * LOG2E);  // chunk 0: exp2(-inf) = 0
      mrow[r] = mn;
      rsum[r] = 0.f;
    }
#pragma unroll
    for (int n = 0; n < 4; ++n)
#pragma unroll
      for (int r = 0; r < 8; ++r) {
        const float pv = exp2f((sc[n][r] - mrow[r]) * LOG2E);
        sc[n][r] = pv;
        rsum[r] += pv;
      }
#pragma unroll
    for (int r = 0; r < 8; ++r) {
#pragma unroll
      for (int off = 1; off < 16; off <<= 1)
        rsum[r] += __shfl_xor(rsum[r], off, 16);
      lrow[r] = lrow[r] * corr[r] + rsum[r];
    }
#pragma unroll
    for (int j = 0; j < 8; ++j)
#pragma unroll
      for (int r = 0; r < 8; ++r) acc[j][r] *= corr[r];

    // probs -> per-wave LDS region, re-read as A fragments for P @ V
    __bf16* Pw = ldsP + wave * (16 * 64);
#pragma unroll
    for (int n = 0; n < 4; ++n)
#pragma unroll
      for (int r = 0; r < 8; ++r)
        Pw[(r + rh) * 64 + n * 16 + ln] = (__bf16)sc[n][r];

    const v16bf ap0 = load_a_frag(Pw, 64, 0, 0, lane);
    const v16bf ap1 = load_a_frag(Pw, 64, 0, 32, lane);
#pragma unroll
    for (int j = 0; j < 8; ++j) {
      acc[j] = wmma_bf16(ap0, load_b_frag(ldsVt, AV_LD, j * 16, 0, lane), acc[j]);
      acc[j] = wmma_bf16(ap1, load_b_frag(ldsVt, AV_LD, j * 16, 32, lane), acc[j]);
    }
    __syncthreads();
  }

  // normalize + store bf16 attn output [b*LI + row][h*DH + d]
#pragma unroll
  for (int r = 0; r < 8; ++r) {
    const float inv = 1.0f / lrow[r];
    const int row = qBase + wave * 16 + r + rh;
#pragma unroll
    for (int j = 0; j < 8; ++j)
      attnb[((size_t)b * LI + row) * (HH * DH) + h * DH + j * 16 + ln] =
          (__bf16)(acc[j][r] * inv);
  }
}

// ---------------------------------------------------------------------------
// Host launch
// ---------------------------------------------------------------------------
extern "C" void kernel_launch(void* const* d_in, const int* in_sizes, int n_in,
                              void* d_out, int out_size, void* d_ws,
                              size_t ws_size, hipStream_t stream) {
  (void)in_sizes; (void)n_in; (void)out_size; (void)ws_size;
  const float* hs   = (const float*)d_in[0];
  const float* enc  = (const float*)d_in[1];
  const unsigned char* msk = (const unsigned char*)d_in[2];  // bool8
  const float* fc   = (const float*)d_in[3];
  const float* wqkv = (const float*)d_in[4];
  const float* wkv  = (const float*)d_in[5];
  const float* wout = (const float*)d_in[6];
  const float* gq   = (const float*)d_in[7];
  const float* gk   = (const float*)d_in[8];
  const float* gak  = (const float*)d_in[9];
  float* out = (float*)d_out;

  // Workspace carve (bf16 elements), ~147 MB total
  __bf16* p = (__bf16*)d_ws;
  __bf16* hsb   = p; p += (size_t)BB * LI * DD;          // 8.4M
  __bf16* encb  = p; p += (size_t)BB * LT * DD;          // 2.1M
  __bf16* wqkvb = p; p += (size_t)DD * 3 * DD;           // 12.6M
  __bf16* wkvb  = p; p += (size_t)DD * 2 * DD;           // 8.4M
  __bf16* woutb = p; p += (size_t)DD * DD;               // 4.2M
  __bf16* Qb    = p; p += (size_t)BB * HH * LI * DH;     // 8.4M
  __bf16* Kb    = p; p += (size_t)BB * HH * LK * DH;     // 10.5M
  __bf16* Vb    = p; p += (size_t)BB * HH * LK * DH;     // 10.5M
  __bf16* attnb = p; p += (size_t)BB * LI * HH * DH;     // 8.4M

  auto cast = [&](const float* src, __bf16* dst, size_t n) {
    const int n4 = (int)(n / 4);
    cast_f32_bf16<<<(n4 + 255) / 256, 256, 0, stream>>>(src, dst, n4);
  };
  cast(hs,   hsb,   (size_t)BB * LI * DD);
  cast(enc,  encb,  (size_t)BB * LT * DD);
  cast(wqkv, wqkvb, (size_t)DD * 3 * DD);
  cast(wkv,  wkvb,  (size_t)DD * 2 * DD);
  cast(wout, woutb, (size_t)DD * DD);

  // img QKV projection + RMSNorm(q,k) + RoPE -> Q / K(rows 512..) / V
  gemm_kernel<0><<<dim3((BB * LI) / GBM, (3 * DD) / GBN), 256, 0, stream>>>(
      hsb, wqkvb, 3 * DD, Qb, Kb, Vb, fc, gq, gk, nullptr);
  // txt KV projection + RMSNorm(k) -> K(rows 0..512) / V
  gemm_kernel<1><<<dim3((BB * LT) / GBM, (2 * DD) / GBN), 256, 0, stream>>>(
      encb, wkvb, 2 * DD, nullptr, Kb, Vb, nullptr, gak, nullptr, nullptr);
  // flash attention over joint KV (2560), masked text positions
  attn_kernel<<<dim3(LI / 128, BB * HH), 256, 0, stream>>>(Qb, Kb, Vb, msk,
                                                           attnb);
  // out projection -> fp32 output
  gemm_kernel<2><<<dim3((BB * LI) / GBM, DD / GBN), 256, 0, stream>>>(
      attnb, woutb, DD, nullptr, nullptr, nullptr, nullptr, nullptr, nullptr,
      out);
}
